// HyenaAttention_29463475651341
// MI455X (gfx1250) — matrix operand order repaired and checked by reading, not compiled, not run on hardware
//
#include <hip/hip_runtime.h>

// ---------------------------------------------------------------------------
// Hyena attention for MI455X (gfx1250): WMMA f16 GEMMs + LDS FFT convolution
//   B=4, N=4096, D=1024, ORDER=2, FFT_N=4096, FREQ=2049
//
// All GEMMs run on v_wmma_f32_16x16x32_f16 (f32 accumulate). The whole
// intermediate chain (values, proj, v) is kept in f16 to halve HBM traffic
// (23.3 TB/s roofline) and keep proj ~L2-resident (192 MB L2). Weights are
// pre-transposed to [Nout][K] f16 so each B fragment is 32 contiguous bytes
// per lane (2x global_load_b128, L2-resident). A tiles are staged to LDS with
// 128-bit copies; fragments are contiguous 16B runs (ds_load_b128).
// Wave tile 32x64 -> 8 WMMA per K-step.
// ---------------------------------------------------------------------------

typedef __attribute__((ext_vector_type(16))) _Float16 v16h;
typedef __attribute__((ext_vector_type(8)))  _Float16 h8;
typedef __attribute__((ext_vector_type(8)))  float    v8f;

#define HY_B    4
#define HY_N    4096
#define HY_D    1024
#define HY_FREQ 2049
#define FFT_N   4096
#define LOG2N   12
#define PI_F    3.14159265358979323846f

// ---------------------------------------------------------------------------
// Plain f32 -> f16 conversion (for values)
// ---------------------------------------------------------------------------
__global__ __launch_bounds__(256) void hy_f32_to_f16(
    const float* __restrict__ src, _Float16* __restrict__ dst, int n)
{
    const int i = (blockIdx.x * 256 + threadIdx.x) * 4;
    if (i < n) {
        const float4 v = *(const float4*)&src[i];
        _Float16* d = &dst[i];
        d[0] = (_Float16)v.x; d[1] = (_Float16)v.y;
        d[2] = (_Float16)v.z; d[3] = (_Float16)v.w;
    }
}

// ---------------------------------------------------------------------------
// Transposing f32 -> f16 weight conversion: dst[n][k] = (f16)src[k][n]
// ---------------------------------------------------------------------------
__global__ __launch_bounds__(256) void hy_convert_transpose(
    const float* __restrict__ src, _Float16* __restrict__ dst, int K, int Nout)
{
    __shared__ _Float16 t[32][33];
    const int n0 = blockIdx.x * 32;
    const int k0 = blockIdx.y * 32;
    const int tx = threadIdx.x & 31;
    const int ty = threadIdx.x >> 5;     // 8 rows of 32
    for (int r = ty; r < 32; r += 8)
        t[r][tx] = (_Float16)src[(size_t)(k0 + r) * Nout + n0 + tx];
    __syncthreads();
    for (int r = ty; r < 32; r += 8)
        dst[(size_t)(n0 + r) * K + k0 + tx] = t[tx][r];
}

// ---------------------------------------------------------------------------
// Tiled WMMA GEMM:  C[M x Nout] = A[M x K](f16, row stride lda) * WT^T
//   WT: [Nout x K] f16, K contiguous.
// mode 0:   C = acc + bias
// mode 1:   C = Vprev * sigmoid(acc + bias)     (fused Hyena gate)
// outF32=1: C is f32 (final output), else f16.
// Block tile 128(M) x 128(N), K step 32; 8 waves; wave tile 32x64.
// ---------------------------------------------------------------------------
#define BM 128
#define BN 128
#define BK 32
#define LDA_SH 40    // 32 + 8 pad (f16 elems); 80B row stride (16B aligned)

__global__ __launch_bounds__(256) void hy_wmma_gemm(
    const _Float16* __restrict__ A, int lda,
    const _Float16* __restrict__ WT,
    const float* __restrict__ bias,
    void* __restrict__ Cout, int ldc,
    const _Float16* __restrict__ Vprev, int ldv,
    int K, int mode, int outF32)
{
    __shared__ __attribute__((aligned(16))) _Float16 Ash[BM * LDA_SH];

    const int tid   = threadIdx.x;
    const int lane  = tid & 31;
    const int wave  = tid >> 5;
    const int waveM = wave >> 1;       // 0..3  (32 rows each)
    const int waveN = wave & 1;        // 0..1  (64 cols each)
    const int bm    = blockIdx.x * BM;
    const int bn    = blockIdx.y * BN;

    v8f acc[2][4] = {};

    // Per-lane B pointers. B fragment (32x16 f16): lanes 0-15 hold K=0..15 of
    // column n=lane; lanes 16-31 hold K=16..31. With WT[n][k] layout each
    // lane's 16 values are contiguous (32B).
    const int colLane = bn + (waveN << 6) + (lane & 15);
    const int kb2     = (lane >> 4) << 4;     // 0 or 16
    const _Float16* bptr[4];
    #pragma unroll
    for (int t = 0; t < 4; ++t)
        bptr[t] = WT + (size_t)(colLane + (t << 4)) * K + kb2;

    // A fragment addressing: lanes 0-15: row=lane,    K 0-15
    //                        lanes 16-31: row=lane-16, K 16-31
    const int arow = (waveM << 5) + (lane & 15);
    const int kbA  = (lane >> 4) << 4;        // 0 or 16

    for (int k0 = 0; k0 < K; k0 += BK) {
        // Prefetch next A tile (emits global_prefetch_b8).
        if (k0 + BK < K)
            __builtin_prefetch(&A[(size_t)(bm + tid) * lda + k0 + BK], 0, 0);

        // Stage A tile (128x32 f16) with 128-bit copies.
        for (int seg = tid; seg < (BM * BK) / 8; seg += 256) {
            const int r = seg >> 2;           // 4 segs of 8 per row
            const int c = (seg & 3) << 3;
            *(h8*)&Ash[r * LDA_SH + c] =
                *(const h8*)&A[(size_t)(bm + r) * lda + k0 + c];
        }
        __syncthreads();

        // Two A fragments (rows m0, m0+16): contiguous 16B runs in LDS.
        v16h af[2];
        #pragma unroll
        for (int s = 0; s < 2; ++s) {
            const _Float16* ap = &Ash[(arow + (s << 4)) * LDA_SH];
            #pragma unroll
            for (int j = 0; j < 8; ++j) {
                af[s][j]     = ap[kbA + j];
                af[s][j + 8] = ap[kbA + 8 + j];
            }
        }

        // 4 B fragments direct from global (L2-resident), 8 WMMAs.
        #pragma unroll
        for (int t = 0; t < 4; ++t) {
            const v16h bf = *(const v16h*)(bptr[t] + k0);
            #pragma unroll
            for (int s = 0; s < 2; ++s)
                acc[s][t] = __builtin_amdgcn_wmma_f32_16x16x32_f16(
                    false, af[s], false, bf, (short)0, acc[s][t], false, false);
        }
        __syncthreads();
    }

    // Epilogue. C/D layout: VGPR r, lanes 0-15 -> M=r, lanes 16-31 -> M=8+r.
    const int colB = bn + (waveN << 6) + (lane & 15);
    #pragma unroll
    for (int s = 0; s < 2; ++s) {
        const int rowBase = bm + (waveM << 5) + (s << 4) + ((lane >> 4) << 3);
        #pragma unroll
        for (int t = 0; t < 4; ++t) {
            const int gn = colB + (t << 4);
            const float bsv = bias[gn];
            #pragma unroll
            for (int r = 0; r < 8; ++r) {
                const int gm = rowBase + r;
                float val = acc[s][t][r] + bsv;
                if (mode == 1) {
                    const float g = 1.0f / (1.0f + __expf(-val));
                    val = (float)Vprev[(size_t)gm * ldv + gn] * g;
                }
                if (outF32) ((float*)Cout)[(size_t)gm * ldc + gn] = val;
                else        ((_Float16*)Cout)[(size_t)gm * ldc + gn] = (_Float16)val;
            }
        }
    }
}

// ---------------------------------------------------------------------------
// Fused per-channel circular convolution: v <- irfft(rfft(v) * H), in place.
// v is f16 (bandwidth-bound stage); FFT computed in f32 in LDS.
// One block = 2 channels of one batch; FFT length 4096, complex f32 in LDS.
// Forward DIF (natural -> bit-reversed), pointwise multiply using __brev to
// find the spectral index (Hermitian-extend the 2049-bin filter), inverse DIT.
// ---------------------------------------------------------------------------
#define CPB 2   // channels per block -> 2 * 4096 * 8B = 64 KB LDS

__global__ __launch_bounds__(256) void hy_fft_conv(
    _Float16* __restrict__ v,          // (B, N, D) f16, in place
    const float* __restrict__ filt)    // (D, FREQ, 2) f32 for this order
{
    __shared__ float2 buf[CPB][FFT_N];

    const int tid     = threadIdx.x;
    const int dGroups = HY_D / CPB;
    const int b       = blockIdx.x / dGroups;
    const int d0      = (blockIdx.x % dGroups) * CPB;
    _Float16* vb = v + (size_t)b * HY_N * HY_D;

    // Cooperative load: buf[dd][n] = v[b][n][d0+dd]  (f16 -> complex f32)
    for (int idx = tid; idx < CPB * FFT_N; idx += 256) {
        const int dd = idx & (CPB - 1);
        const int n  = idx >> 1;
        buf[dd][n] = make_float2((float)vb[(size_t)n * HY_D + d0 + dd], 0.0f);
    }
    __syncthreads();

    const int ch = tid >> 7;        // 128 threads per channel
    const int lt = tid & 127;
    float2* x = buf[ch];

    // Forward DIF
    for (int half = FFT_N >> 1, sh = LOG2N - 1; half >= 1; half >>= 1, --sh) {
        for (int t = lt; t < (FFT_N >> 1); t += 128) {
            const int j  = t & (half - 1);
            const int i0 = ((t >> sh) << (sh + 1)) + j;
            const int i1 = i0 + half;
            const float2 u = x[i0];
            const float2 w = x[i1];
            const float2 dif = make_float2(u.x - w.x, u.y - w.y);
            float sn, cs;
            __sincosf(-PI_F * (float)j / (float)half, &sn, &cs);
            x[i0] = make_float2(u.x + w.x, u.y + w.y);
            x[i1] = make_float2(dif.x * cs - dif.y * sn, dif.x * sn + dif.y * cs);
        }
        __syncthreads();
    }

    // Pointwise multiply by Hermitian-extended filter; fold in 1/N scale.
    for (int idx = tid; idx < CPB * FFT_N; idx += 256) {
        const int dd = idx & (CPB - 1);
        const int p  = idx >> 1;
        const int k  = (int)(__brev((unsigned)p) >> (32 - LOG2N)); // spectrum idx
        const float* f = filt + (size_t)(d0 + dd) * HY_FREQ * 2;
        float hr, hi;
        if (k <= FFT_N / 2) { hr = f[2 * k];             hi =  f[2 * k + 1]; }
        else                { hr = f[2 * (FFT_N - k)];   hi = -f[2 * (FFT_N - k) + 1]; }
        const float2 a = buf[dd][p];
        const float s = 1.0f / (float)FFT_N;
        buf[dd][p] = make_float2((a.x * hr - a.y * hi) * s,
                                 (a.x * hi + a.y * hr) * s);
    }
    __syncthreads();

    // Inverse DIT (bit-reversed -> natural)
    for (int half = 1, sh = 0; half < FFT_N; half <<= 1, ++sh) {
        for (int t = lt; t < (FFT_N >> 1); t += 128) {
            const int j  = t & (half - 1);
            const int i0 = ((t >> sh) << (sh + 1)) + j;
            const int i1 = i0 + half;
            const float2 u = x[i0];
            const float2 w = x[i1];
            float sn, cs;
            __sincosf(PI_F * (float)j / (float)half, &sn, &cs);
            const float2 tw = make_float2(w.x * cs - w.y * sn, w.x * sn + w.y * cs);
            x[i0] = make_float2(u.x + tw.x, u.y + tw.y);
            x[i1] = make_float2(u.x - tw.x, u.y - tw.y);
        }
        __syncthreads();
    }

    // Store real part back (f32 -> f16).
    for (int idx = tid; idx < CPB * FFT_N; idx += 256) {
        const int dd = idx & (CPB - 1);
        const int n  = idx >> 1;
        vb[(size_t)n * HY_D + d0 + dd] = (_Float16)buf[dd][n].x;
    }
}

// ---------------------------------------------------------------------------
// Launch pipeline
// ---------------------------------------------------------------------------
extern "C" void kernel_launch(void* const* d_in, const int* in_sizes, int n_in,
                              void* d_out, int out_size, void* d_ws, size_t ws_size,
                              hipStream_t stream) {
    (void)in_sizes; (void)n_in; (void)out_size; (void)ws_size;

    const float* values     = (const float*)d_in[2];
    const float* in_proj_w  = (const float*)d_in[3];
    const float* in_proj_b  = (const float*)d_in[4];
    const float* out_proj_w = (const float*)d_in[5];
    const float* out_proj_b = (const float*)d_in[6];
    const float* gate_w     = (const float*)d_in[7];
    const float* gate_b     = (const float*)d_in[8];
    const float* filters    = (const float*)d_in[9];

    const int M  = HY_B * HY_N;          // 16384
    const int K  = HY_D;                 // 1024
    const int D3 = 3 * HY_D;             // 3072

    // Workspace carve-up (all f16 intermediates)
    _Float16* ws = (_Float16*)d_ws;
    _Float16* values16  = ws;                                   // M * D
    _Float16* proj16    = values16 + (size_t)M * HY_D;          // M * 3D
    _Float16* vA16      = proj16 + (size_t)M * D3;              // M * D
    _Float16* vB16      = vA16 + (size_t)M * HY_D;              // M * D
    _Float16* w16t_in   = vB16 + (size_t)M * HY_D;              // 3D * D (transposed)
    _Float16* w16t_gate = w16t_in + (size_t)HY_D * D3;          // 2 * (D * D)
    _Float16* w16t_out  = w16t_gate + (size_t)2 * HY_D * HY_D;  // D * D

    const dim3 blk(256);

    // 1) values -> f16; weights -> f16 transposed WT[n][k]
    const int nv = M * HY_D;
    hy_f32_to_f16<<<dim3(nv / (256 * 4)), blk, 0, stream>>>(values, values16, nv);
    hy_convert_transpose<<<dim3(D3 / 32, HY_D / 32), blk, 0, stream>>>(
        in_proj_w, w16t_in, K, D3);
    hy_convert_transpose<<<dim3(HY_D / 32, HY_D / 32), blk, 0, stream>>>(
        gate_w, w16t_gate, K, HY_D);
    hy_convert_transpose<<<dim3(HY_D / 32, HY_D / 32), blk, 0, stream>>>(
        gate_w + (size_t)HY_D * HY_D, w16t_gate + (size_t)HY_D * HY_D, K, HY_D);
    hy_convert_transpose<<<dim3(HY_D / 32, HY_D / 32), blk, 0, stream>>>(
        out_proj_w, w16t_out, K, HY_D);

    // 2) proj = values @ in_proj_w + b          (16384x1024 * 1024x3072)
    hy_wmma_gemm<<<dim3(M / BM, D3 / BN), blk, 0, stream>>>(
        values16, HY_D, w16t_in, in_proj_b, proj16, D3, nullptr, 0, K, 0, 0);

    // 3) order 0: vA = proj[:,:,0] * sigmoid(proj[:,:,1] @ gate_w[0] + b0)
    hy_wmma_gemm<<<dim3(M / BM, HY_D / BN), blk, 0, stream>>>(
        proj16 + HY_D, D3, w16t_gate, gate_b, vA16, HY_D, proj16, D3, K, 1, 0);

    //    vA <- irfft(rfft(vA) * H0)
    hy_fft_conv<<<dim3(HY_B * (HY_D / CPB)), blk, 0, stream>>>(vA16, filters);

    // 4) order 1: vB = vA * sigmoid(proj[:,:,2] @ gate_w[1] + b1)
    hy_wmma_gemm<<<dim3(M / BM, HY_D / BN), blk, 0, stream>>>(
        proj16 + 2 * HY_D, D3, w16t_gate + (size_t)HY_D * HY_D, gate_b + HY_D,
        vB16, HY_D, vA16, HY_D, K, 1, 0);

    //    vB <- irfft(rfft(vB) * H1)
    hy_fft_conv<<<dim3(HY_B * (HY_D / CPB)), blk, 0, stream>>>(
        vB16, filters + (size_t)HY_D * HY_FREQ * 2);

    // 5) out = vB @ out_proj_w + b   (f32 output)
    hy_wmma_gemm<<<dim3(M / BM, HY_D / BN), blk, 0, stream>>>(
        vB16, HY_D, w16t_out, out_proj_b, d_out, HY_D, nullptr, 0, K, 0, 1);
}
